// ECSAttention_22187801051589
// MI455X (gfx1250) — matrix operand order repaired and checked by reading, not compiled, not run on hardware
//
#include <hip/hip_runtime.h>
#include <hip/hip_bf16.h>
#include <stdint.h>

typedef __bf16 bf16;
typedef __attribute__((ext_vector_type(16))) __bf16 v16bf;
typedef __attribute__((ext_vector_type(8)))  float  v8f;
typedef unsigned int u32x4 __attribute__((ext_vector_type(4)));
typedef int i32x8 __attribute__((ext_vector_type(8)));
typedef int i32x4 __attribute__((ext_vector_type(4)));

union V16 { v16bf v; uint4 u[2]; };
union PK8 { uint4 u; bf16 h[8]; };

#ifdef __has_builtin
#if __has_builtin(__builtin_amdgcn_tensor_load_to_lds) && __has_builtin(__builtin_amdgcn_s_wait_tensorcnt)
#define USE_TDM 1
#endif
#endif
#ifndef USE_TDM
#define USE_TDM 0
#endif

#if USE_TDM
// Tensor DMA descriptor per CDNA5 ISA ch.8: 2D tile (tile_rows x tile_row_bytes),
// data_size=1B, global row stride row_stride_bytes, rows >= tensor_rows zero-filled,
// LDS padding pad_amount after every pad_interval (codes per D# group1 table).
__device__ __forceinline__ void tdm_load_tile(
    const void* gsrc, unsigned lds_addr,
    unsigned tile_row_bytes, unsigned tile_rows,
    unsigned long long row_stride_bytes, unsigned tensor_rows,
    unsigned pad_interval_code, unsigned pad_amount_code)
{
  unsigned long long ga = (unsigned long long)(uintptr_t)gsrc;
  u32x4 g0;
  g0.x = 1u;                                                   // count=1 (valid), user mode
  g0.y = lds_addr;                                             // lds_addr [63:32]
  g0.z = (unsigned)ga;                                         // global_addr [95:64]
  g0.w = (unsigned)((ga >> 32) & 0x01ffffffu) | (2u << 30);    // global_addr hi + type=2
  unsigned dim0 = (unsigned)row_stride_bytes;                  // tensor_dim0 (bytes)
  unsigned dim1 = tensor_rows;                                 // tensor_dim1
  i32x8 g1;
  g1[0] = (int)((1u << 20)                                     // pad_enable
              | (pad_interval_code << 22)
              | (pad_amount_code << 25));
  g1[1] = (int)((dim0 & 0xffffu) << 16);                       // atomic_addr=0 | dim0[15:0]
  g1[2] = (int)((dim0 >> 16) | ((dim1 & 0xffffu) << 16));      // dim0[31:16] | dim1[15:0]
  g1[3] = (int)((dim1 >> 16) | (tile_row_bytes << 16));        // dim1[31:16] | tile_dim0
  g1[4] = (int)(tile_rows & 0xffffu);                          // tile_dim1 | tile_dim2=0
  g1[5] = (int)(unsigned)(row_stride_bytes & 0xffffffffu);     // tensor_dim0_stride[31:0]
  g1[6] = (int)(unsigned)((row_stride_bytes >> 32) & 0xffffu); // stride[47:32] | dim1_stride=0
  g1[7] = 0;
  i32x4 z4 = {0, 0, 0, 0};
#if defined(__clang_major__) && __clang_major__ >= 23
  i32x8 z8 = {0, 0, 0, 0, 0, 0, 0, 0};
  __builtin_amdgcn_tensor_load_to_lds(g0, g1, z4, z4, z8, 0);
#else
  __builtin_amdgcn_tensor_load_to_lds(g0, g1, z4, z4, 0);
#endif
}
#endif

// ---- WMMA fragment loaders (layouts per CDNA5 ISA 7.12.2, wave32) ----
__device__ __forceinline__ v16bf load_a_frag(const bf16* A, int lane, int row0, int k0, int ld) {
  const bf16* p = A + (size_t)(row0 + (lane & 15)) * ld + k0 + ((lane & 16) ? 8 : 0);
  V16 r;
  r.u[0] = *(const uint4*)p;          // K offsets 0..7
  r.u[1] = *(const uint4*)(p + 16);   // K offsets 16..23
  return r.v;
}
__device__ __forceinline__ v16bf load_b_frag(const bf16* B, int lane, int n0, int k0, int ld) {
  const bf16* p = B + (size_t)(n0 + (lane & 15)) * ld + k0 + ((lane & 16) ? 8 : 0);
  V16 r;
  r.u[0] = *(const uint4*)p;
  r.u[1] = *(const uint4*)(p + 16);
  return r.v;
}
#define WMMA_BF16(a,b,c) __builtin_amdgcn_wmma_f32_16x16x32_bf16(false,(a),false,(b),(short)0,(c),false,false)

__device__ __forceinline__ v8f zero8() { v8f z = {0.f,0.f,0.f,0.f,0.f,0.f,0.f,0.f}; return z; }

// ---------------- small helper kernels ----------------
__global__ void k_cvt(const float* __restrict__ in, bf16* __restrict__ outp, int n) {
  int i = blockIdx.x * 256 + threadIdx.x;
  if (i < n) outp[i] = (bf16)in[i];
}

__global__ void k_w5(const float* __restrict__ dw1, const float* __restrict__ dw2, bf16* __restrict__ W5) {
  int i = blockIdx.x * 256 + threadIdx.x;
  if (i >= 5 * 64 * 64) return;
  int t = i / 4096, rem = i % 4096;
  int f = rem >> 6, e = rem & 63;
  float v = dw2[(f * 64 + e) * 5 + t];
  if (t >= 1 && t <= 3) v += dw1[(f * 64 + e) * 3 + (t - 1)];
  W5[t * 4096 + f * 64 + e] = (bf16)v;
}

__global__ void k_gate(const bf16* __restrict__ V, const float* __restrict__ conv_w,
                       const float* __restrict__ conv_b, float* __restrict__ res) {
  __shared__ float mean[64];
  const int bh = blockIdx.x, t = threadIdx.x;
  const bf16* vp = V + (size_t)bh * 4096 * 64;
  float s = 0.f;
  for (int n = 0; n < 4096; ++n) s += (float)vp[(size_t)n * 64 + t];
  mean[t] = s * (1.0f / 4096.0f);
  __syncthreads();
  float acc = conv_b[t];
  #pragma unroll 8
  for (int ci = 0; ci < 64; ++ci) acc += conv_w[(t * 64 + ci) * 5 + 2] * mean[ci];
  res[bh * 64 + t] = 1.f / (1.f + __expf(-acc));
}

// ---------------- unified bf16 WMMA GEMM (128x128x32 tiles, TDM-fed LDS) ----------------
template<int EPI>
__global__ __launch_bounds__(256) void k_gemm(
    const bf16* __restrict__ Ain, const bf16* __restrict__ Bw,
    bf16* __restrict__ o0, bf16* __restrict__ o1, bf16* __restrict__ o2,
    const float* __restrict__ bias, float* __restrict__ outf)
{
  __shared__ bf16 As[2][128 * 40];   // 64B data + 16B pad per row
  __shared__ bf16 Bs[2][128 * 40];
  const int tid = threadIdx.x, lane = tid & 31, wid = tid >> 5;
  const int mw = (wid & 3) * 32, nw = (wid >> 2) * 64;
  const int m0 = blockIdx.y * 128, n0b = blockIdx.x * 128;

  auto fetch = [&](int kt, int buf) {
#if USE_TDM
    if (wid == 0)
      tdm_load_tile(Ain + (size_t)m0 * 768 + kt * 32, (unsigned)(uintptr_t)&As[buf][0],
                    64u, 128u, 1536ull, 128u, 3u, 3u);
    else if (wid == 1)
      tdm_load_tile(Bw + (size_t)n0b * 768 + kt * 32, (unsigned)(uintptr_t)&Bs[buf][0],
                    64u, 128u, 1536ull, 128u, 3u, 3u);
#else
    #pragma unroll
    for (int q = 0; q < 2; ++q) {
      int c = tid + q * 256;
      int row = c >> 2, cc = (c & 3) * 8;
      *(uint4*)&As[buf][row * 40 + cc] = *(const uint4*)(Ain + (size_t)(m0 + row) * 768 + kt * 32 + cc);
      *(uint4*)&Bs[buf][row * 40 + cc] = *(const uint4*)(Bw + (size_t)(n0b + row) * 768 + kt * 32 + cc);
      __builtin_prefetch(Ain + (size_t)(m0 + row) * 768 + kt * 32 + cc + 32, 0, 1);
    }
#endif
  };
  auto fetch_wait = [&]() {
#if USE_TDM
    if (wid < 2) __builtin_amdgcn_s_wait_tensorcnt(0);
#endif
    __syncthreads();
  };

  v8f acc[2][4];
  #pragma unroll
  for (int mi = 0; mi < 2; ++mi)
    #pragma unroll
    for (int j = 0; j < 4; ++j) acc[mi][j] = zero8();

  fetch(0, 0);
  fetch_wait();
  for (int kt = 0; kt < 24; ++kt) {
    int cur = kt & 1;
    if (kt + 1 < 24) fetch(kt + 1, 1 - cur);   // DMA next tile while computing
    v16bf bfr[4];
    #pragma unroll
    for (int j = 0; j < 4; ++j) bfr[j] = load_b_frag(&Bs[cur][0], lane, nw + 16 * j, 0, 40);
    #pragma unroll
    for (int mi = 0; mi < 2; ++mi) {
      v16bf af = load_a_frag(&As[cur][0], lane, mw + 16 * mi, 0, 40);
      #pragma unroll
      for (int j = 0; j < 4; ++j) acc[mi][j] = WMMA_BF16(af, bfr[j], acc[mi][j]);
    }
    if (kt + 1 < 24) fetch_wait();
  }

  #pragma unroll
  for (int mi = 0; mi < 2; ++mi) {
    int row0 = m0 + mw + 16 * mi + ((lane & 16) ? 8 : 0);
    #pragma unroll
    for (int j = 0; j < 4; ++j) {
      int col = n0b + nw + 16 * j + (lane & 15);
      if constexpr (EPI == 0) {
        int b = row0 >> 12, n = row0 & 4095;
        int s = col / 768, c = col % 768;
        int h = c >> 6, dd = c & 63;
        if (s == 0) {
          PK8 pk;
          #pragma unroll
          for (int r = 0; r < 8; ++r) pk.h[r] = (bf16)acc[mi][j][r];
          *(uint4*)(o0 + ((size_t)((b * 12 + h) * 64 + dd)) * 4096 + n) = pk.u;
        } else if (s == 1) {
          PK8 pk;
          #pragma unroll
          for (int r = 0; r < 8; ++r) pk.h[r] = (bf16)(acc[mi][j][r] * 0.125f);  // d^-0.5
          *(uint4*)(o1 + ((size_t)((b * 12 + h) * 64 + dd)) * 4096 + n) = pk.u;
        } else {
          #pragma unroll
          for (int r = 0; r < 8; ++r)
            o2[((size_t)(b * 12 + h) * 4096 + n + r) * 64 + dd] = (bf16)acc[mi][j][r];
        }
      } else {
        float bb = bias[col];
        #pragma unroll
        for (int r = 0; r < 8; ++r)
          outf[(size_t)(row0 + r) * 768 + col] = acc[mi][j][r] + bb;
      }
    }
  }
}

// ---------------- per-head: kq = (k*scale)^T q -> relu,fc,softmax -> a (bf16) ----------------
__global__ __launch_bounds__(128) void k_kq(const bf16* __restrict__ kT, const bf16* __restrict__ qT,
                                            const float* __restrict__ fc_w, const float* __restrict__ fc_b,
                                            bf16* __restrict__ abuf)
{
  __shared__ float kqs[64 * 64];
  __shared__ float as2[64 * 64];
  __shared__ float fcs[64 * 64];
  const int bh = blockIdx.x;
  const int tid = threadIdx.x, lane = tid & 31, w = tid >> 5;
  const bf16* kp = kT + (size_t)bh * 64 * 4096;
  const bf16* qp = qT + (size_t)bh * 64 * 4096;
  for (int i = tid; i < 4096; i += 128) fcs[i] = fc_w[i];

  v8f acc[4] = {zero8(), zero8(), zero8(), zero8()};
  for (int n0 = 0; n0 < 4096; n0 += 32) {
    v16bf af = load_a_frag(kp, lane, 16 * w, n0, 4096);
    #pragma unroll
    for (int j = 0; j < 4; ++j) {
      v16bf bfr = load_b_frag(qp, lane, 16 * j, n0, 4096);
      acc[j] = WMMA_BF16(af, bfr, acc[j]);
    }
  }
  {
    int rr = 16 * w + ((lane & 16) ? 8 : 0);
    #pragma unroll
    for (int j = 0; j < 4; ++j) {
      int cc = 16 * j + (lane & 15);
      #pragma unroll
      for (int r = 0; r < 8; ++r) kqs[(rr + r) * 64 + cc] = acc[j][r];
    }
  }
  __syncthreads();
  for (int idx = tid; idx < 4096; idx += 128) {
    int e = idx >> 6, f = idx & 63;
    float s = fc_b[f] + kqs[e * 64 + f];
    const float* kr = &kqs[e * 64];
    #pragma unroll 8
    for (int g = 0; g < 64; ++g) s += fmaxf(kr[g], 0.f) * fcs[f * 64 + g];
    as2[idx] = s;
  }
  __syncthreads();
  if (tid < 64) {
    const float* rowp = &as2[tid * 64];
    float m = rowp[0];
    for (int g = 1; g < 64; ++g) m = fmaxf(m, rowp[g]);
    float sum = 0.f;
    for (int g = 0; g < 64; ++g) sum += __expf(rowp[g] - m);
    float inv = 1.f / sum;
    bf16* ap = abuf + (size_t)bh * 4096 + tid * 64;
    for (int g = 0; g < 64; ++g) ap[g] = (bf16)(__expf(rowp[g] - m) * inv);
  }
}

// ---------------- per-head token tile: conv (5-tap WMMA) -> x a^T -> gate -> Y ----------------
__global__ __launch_bounds__(256) void k_conv_av(
    const bf16* __restrict__ V, const bf16* __restrict__ W5, const bf16* __restrict__ abuf,
    const float* __restrict__ res, const float* __restrict__ b1, const float* __restrict__ b2,
    bf16* __restrict__ Y)
{
  __shared__ bf16 vs[132 * 72];    // tokens n0-2 .. n0+129; 128B data + 16B pad per row
  __shared__ bf16 v2s[128 * 72];
  const int bh = blockIdx.y, b = bh / 12, h = bh % 12;
  const int n0 = blockIdx.x * 128;
  const int tid = threadIdx.x, lane = tid & 31, w = tid >> 5;
  const bf16* vp = V + (size_t)bh * 4096 * 64;

#if USE_TDM
  if (n0 == 0 && tid < 18) {   // zero-fill 2 left-halo rows (144 bf16 = 18 x 16B)
    uint4 z = make_uint4(0u, 0u, 0u, 0u);
    *(uint4*)&vs[tid * 8] = z;
  }
  if (w == 0) {
    int gs = (n0 == 0) ? 0 : n0 - 2;
    int row_off = (n0 == 0) ? 2 : 0;
    tdm_load_tile(vp + (size_t)gs * 64, (unsigned)(uintptr_t)&vs[row_off * 72],
                  128u, (unsigned)(132 - row_off), 128ull, (unsigned)(4096 - gs), 4u, 3u);
    __builtin_amdgcn_s_wait_tensorcnt(0);
  }
  __syncthreads();
#else
  for (int c = tid; c < 132 * 8; c += 256) {
    int row = c >> 3, cc = (c & 7) * 8;
    int n = n0 + row - 2;
    uint4 val = make_uint4(0u, 0u, 0u, 0u);
    if (n >= 0 && n < 4096) val = *(const uint4*)(vp + (size_t)n * 64 + cc);
    *(uint4*)&vs[row * 72 + cc] = val;
  }
  __syncthreads();
#endif

  v8f acc[4] = {zero8(), zero8(), zero8(), zero8()};
  #pragma unroll
  for (int t = 0; t < 5; ++t)
    #pragma unroll
    for (int kk = 0; kk < 2; ++kk) {
      v16bf af = load_a_frag(vs, lane, 16 * w + t, 32 * kk, 72);
      #pragma unroll
      for (int j = 0; j < 4; ++j) {
        v16bf bfr = load_b_frag(W5 + t * 4096, lane, 16 * j, 32 * kk, 64);
        acc[j] = WMMA_BF16(af, bfr, acc[j]);
      }
    }
  {
    int rr = 16 * w + ((lane & 16) ? 8 : 0);
    #pragma unroll
    for (int j = 0; j < 4; ++j) {
      int f = 16 * j + (lane & 15);
      float bias = b1[f] + b2[f];
      #pragma unroll
      for (int r = 0; r < 8; ++r) v2s[(rr + r) * 72 + f] = (bf16)(acc[j][r] + bias);
    }
  }
  __syncthreads();

  v8f oacc[4] = {zero8(), zero8(), zero8(), zero8()};
  const bf16* ap = abuf + (size_t)bh * 4096;
  #pragma unroll
  for (int kk = 0; kk < 2; ++kk) {
    v16bf af = load_a_frag(v2s, lane, 16 * w, 32 * kk, 72);
    #pragma unroll
    for (int j = 0; j < 4; ++j) {
      v16bf bfr = load_b_frag(ap, lane, 16 * j, 32 * kk, 64);
      oacc[j] = WMMA_BF16(af, bfr, oacc[j]);
    }
  }
  {
    int row0 = n0 + 16 * w + ((lane & 16) ? 8 : 0);
    #pragma unroll
    for (int j = 0; j < 4; ++j) {
      int g = 16 * j + (lane & 15);
      float gate = res[bh * 64 + g];
      bf16* yp = Y + (size_t)(b * 4096 + row0) * 768 + h * 64 + g;
      #pragma unroll
      for (int r = 0; r < 8; ++r) yp[(size_t)r * 768] = (bf16)(oacc[j][r] * gate);
    }
  }
}

// ---------------- host ----------------
extern "C" void kernel_launch(void* const* d_in, const int* in_sizes, int n_in,
                              void* d_out, int out_size, void* d_ws, size_t ws_size,
                              hipStream_t stream) {
  (void)in_sizes; (void)n_in; (void)out_size; (void)ws_size;
  const float* x      = (const float*)d_in[0];
  const float* qkv_w  = (const float*)d_in[1];
  const float* proj_w = (const float*)d_in[2];
  const float* proj_b = (const float*)d_in[3];
  const float* conv_w = (const float*)d_in[4];
  const float* conv_b = (const float*)d_in[5];
  const float* fc_w   = (const float*)d_in[6];
  const float* fc_b   = (const float*)d_in[7];
  const float* dw1_w  = (const float*)d_in[8];
  const float* dw1_b  = (const float*)d_in[9];
  const float* dw2_w  = (const float*)d_in[10];
  const float* dw2_b  = (const float*)d_in[11];
  float* outp = (float*)d_out;

  char* base = (char*)d_ws; size_t off = 0;
  auto carve = [&](size_t bytes) { void* p = base + off; off = (off + bytes + 255) & ~(size_t)255; return p; };
  bf16*  qkvwB  = (bf16*)carve((size_t)2304 * 768 * 2);
  bf16*  projwB = (bf16*)carve((size_t)768 * 768 * 2);
  bf16*  xbf    = (bf16*)carve((size_t)32768 * 768 * 2);
  bf16*  qT     = (bf16*)carve((size_t)96 * 64 * 4096 * 2);
  bf16*  kT     = (bf16*)carve((size_t)96 * 64 * 4096 * 2);
  bf16*  Vb     = (bf16*)carve((size_t)96 * 4096 * 64 * 2);
  bf16*  W5     = (bf16*)carve((size_t)5 * 64 * 64 * 2);
  float* res    = (float*)carve((size_t)96 * 64 * 4);
  bf16*  abuf   = (bf16*)carve((size_t)96 * 64 * 64 * 2);
  bf16*  Y      = qT;  // alias: qT dead after k_kq; Y written later in stream order

  k_cvt<<<(32768 * 768 + 255) / 256, 256, 0, stream>>>(x, xbf, 32768 * 768);
  k_cvt<<<(2304 * 768 + 255) / 256, 256, 0, stream>>>(qkv_w, qkvwB, 2304 * 768);
  k_cvt<<<(768 * 768 + 255) / 256, 256, 0, stream>>>(proj_w, projwB, 768 * 768);
  k_w5<<<(5 * 4096 + 255) / 256, 256, 0, stream>>>(dw1_w, dw2_w, W5);
  k_gemm<0><<<dim3(18, 256), 256, 0, stream>>>(xbf, qkvwB, qT, kT, Vb, nullptr, nullptr);
  k_gate<<<96, 64, 0, stream>>>(Vb, conv_w, conv_b, res);
  k_kq<<<96, 128, 0, stream>>>(kT, qT, fc_w, fc_b, abuf);
  k_conv_av<<<dim3(32, 96), 256, 0, stream>>>(Vb, W5, abuf, res, dw1_b, dw2_b, Y);
  k_gemm<1><<<dim3(6, 256), 256, 0, stream>>>(Y, projwB, nullptr, nullptr, nullptr, proj_b, outp);
}
